// LinearNO_2765958939227
// MI455X (gfx1250) — matrix operand order repaired and checked by reading, not compiled, not run on hardware
//
#include <hip/hip_runtime.h>
#include <hip/hip_bf16.h>

// ---------------------------------------------------------------------------
// Types for WMMA f16 16x16x32 (CDNA5 / gfx1250, wave32)
// ---------------------------------------------------------------------------
typedef _Float16 v16h __attribute__((ext_vector_type(16)));
typedef _Float16 v8h  __attribute__((ext_vector_type(8)));
typedef float    v8f  __attribute__((ext_vector_type(8)));

struct __align__(8) H4 { _Float16 h[4]; };

__device__ __forceinline__ v16h cat16(v8h lo, v8h hi) {
  return __builtin_shufflevector(lo, hi, 0,1,2,3,4,5,6,7,8,9,10,11,12,13,14,15);
}

__device__ __forceinline__ v8f wmma_f16(v16h a, v16h b, v8f c) {
  // emits v_wmma_f32_16x16x32_f16
  return __builtin_amdgcn_wmma_f32_16x16x32_f16(false, a, false, b, (short)0, c,
                                                false, false);
}

__device__ __forceinline__ v8f vzero8() {
  v8f z = {0.f,0.f,0.f,0.f,0.f,0.f,0.f,0.f};
  return z;
}

// ---------------------------------------------------------------------------
// CDNA5 async global->LDS copy (VGLOBAL GV mode, tracked by ASYNCcnt).
//   global_load_async_to_lds_b128 vdst(lds-offset), vaddr(64b), off
// LDS offset = low 32 bits of the generic shared pointer (addrspacecast on
// AMDGPU is a truncation for generic->LDS).
// ---------------------------------------------------------------------------
__device__ __forceinline__ void async_copy_b128(const _Float16* __restrict__ g,
                                                _Float16* l) {
  unsigned loff = (unsigned)(unsigned long long)l;
  asm volatile("global_load_async_to_lds_b128 %0, %1, off"
               :: "v"(loff), "v"(g) : "memory");
}
__device__ __forceinline__ void wait_async0() {
  asm volatile("s_wait_asynccnt 0x0" ::: "memory");
}

// A fragment (16x32 f16, M x K), source: LDS row-major [m][k], row stride ld.
// Lane L: M = mbase + L%16 ; K = (L/16)*16 .. +15 contiguous -> two ds_load_b128.
__device__ __forceinline__ v16h load_a_lds(const _Float16* lds, int ld,
                                           int mbase, int kk, int lane) {
  const int m    = mbase + (lane & 15);
  const int half = lane >> 4;
  const v8h* p0 = (const v8h*)(lds + m * ld + kk + half * 16);
  const v8h* p1 = (const v8h*)(lds + m * ld + kk + 8 + half * 16);
  return cat16(*p0, *p1);
}

// B fragment (32x16 f16, K x N), source: global weight pre-transposed to
// [n][k] row-major (row stride ldb). Lane L: N = nbase + L%16,
// K = (L/16)*16 .. +15 contiguous -> two global_load_b128.
__device__ __forceinline__ v16h load_b_gbl(const _Float16* __restrict__ Bt,
                                           int ldb, int nbase, int kk, int lane) {
  const int n  = nbase + (lane & 15);
  const int kh = (lane >> 4) * 16;
  const v8h* p = (const v8h*)(Bt + n * ldb + kk + kh);
  return cat16(p[0], p[1]);
}

// ---------------------------------------------------------------------------
// Problem constants
// ---------------------------------------------------------------------------
#define ROWS   131072   // B*N = 8*16384
#define DIMC   512
#define NPERB  16384
#define HEADS  8
#define DH     64

// ---------------------------------------------------------------------------
// Kernel 0: weight transpose/convert to f16 [n][k]; zero kv accumulators
// ---------------------------------------------------------------------------
__global__ __launch_bounds__(256) void prep_k(
    const float* __restrict__ W_in, const float* __restrict__ W_out,
    const float* __restrict__ W_v,  const float* __restrict__ W_k,
    _Float16* __restrict__ Wt_in,  _Float16* __restrict__ Wt_out,
    _Float16* __restrict__ Wt_v,   _Float16* __restrict__ Wt_k,
    float* __restrict__ kv_ws,     float* __restrict__ ksum_ws) {
  const int i = blockIdx.x * 256 + threadIdx.x;
  if (i < 262144) {                       // Wt_in[n][k] = W_in[k][n]
    int n = i >> 9, k = i & 511;
    Wt_in[i] = (_Float16)W_in[k * 512 + n];
  } else if (i < 524288) {                // Wt_out
    int j = i - 262144; int n = j >> 9, k = j & 511;
    Wt_out[j] = (_Float16)W_out[k * 512 + n];
  } else if (i < 528384) {                // Wt_v[c][d] = W_v[d][c]
    int j = i - 524288; int c = j >> 6, d = j & 63;
    Wt_v[j] = (_Float16)W_v[d * 64 + c];
  } else if (i < 529408) {                // Wt_k padded to 16 rows
    int j = i - 528384; int r = j >> 6, d = j & 63;
    Wt_k[j] = (r < 4) ? (_Float16)W_k[d * 4 + r] : (_Float16)0.f;
  } else if (i < 545792) {
    kv_ws[i - 529408] = 0.f;
  } else if (i < 546048) {
    ksum_ws[i - 545792] = 0.f;
  }
}

// ---------------------------------------------------------------------------
// Tiled GEMM: C[rows x 512] = A[rows x 512] @ Bt^T + bias
//   block tile 128x128, 256 threads = 8 waves as 4(m) x 2(n),
//   each wave: 32 rows x 64 cols = 2x4 WMMA tiles, K-step 32.
//   A staged through LDS: f32 source converts in flight (VGPR path),
//   f16 source uses CDNA5 async global->LDS DMA (ASYNCcnt).
// ---------------------------------------------------------------------------
template<bool A_F32, bool OUT_F16>
__global__ __launch_bounds__(256) void gemm512_k(
    const void* __restrict__ Ap, const _Float16* __restrict__ Bt,
    const float* __restrict__ bias, void* __restrict__ Cp) {
  __shared__ _Float16 Alds[128 * 40];   // stride 40 keeps b128 16B-aligned
  const int tid  = threadIdx.x;
  const int lane = tid & 31;
  const int wid  = tid >> 5;
  const int wm   = wid & 3;             // 0..3 -> 32-row slabs
  const int wn   = wid >> 2;            // 0..1 -> 64-col slabs
  const int rowbase = blockIdx.x * 128;
  const int colbase = blockIdx.y * 128;

  v8f acc[2][4];
  for (int mf = 0; mf < 2; ++mf)
    for (int nf = 0; nf < 4; ++nf) acc[mf][nf] = vzero8();

  for (int ks = 0; ks < 512; ks += 32) {
    __syncthreads();
    if (A_F32) {
      const float* A32 = (const float*)Ap;
      #pragma unroll
      for (int i = 0; i < 4; ++i) {         // 1024 float4 loads / block
        int q = tid + i * 256;
        int r = q >> 3, c4 = (q & 7) * 4;
        float4 v = *(const float4*)(A32 + (size_t)(rowbase + r) * 512 + ks + c4);
        H4 hh;
        hh.h[0] = (_Float16)v.x; hh.h[1] = (_Float16)v.y;
        hh.h[2] = (_Float16)v.z; hh.h[3] = (_Float16)v.w;
        *(H4*)&Alds[r * 40 + c4] = hh;
      }
    } else {
      const _Float16* A16 = (const _Float16*)Ap;
      #pragma unroll
      for (int i = 0; i < 2; ++i) {         // 512 async b128 DMAs / block
        int q = tid + i * 256;
        int r = q >> 2, c8 = (q & 3) * 8;
        async_copy_b128(A16 + (size_t)(rowbase + r) * 512 + ks + c8,
                        &Alds[r * 40 + c8]);
      }
      wait_async0();
    }
    __syncthreads();

    v16h af[2], bf[4];
    #pragma unroll
    for (int mf = 0; mf < 2; ++mf)
      af[mf] = load_a_lds(Alds, 40, wm * 32 + mf * 16, 0, lane);
    #pragma unroll
    for (int nf = 0; nf < 4; ++nf)
      bf[nf] = load_b_gbl(Bt, 512, colbase + wn * 64 + nf * 16, ks, lane);
    #pragma unroll
    for (int mf = 0; mf < 2; ++mf)
      #pragma unroll
      for (int nf = 0; nf < 4; ++nf)
        acc[mf][nf] = wmma_f16(af[mf], bf[nf], acc[mf][nf]);
  }

  // Epilogue: C layout lane L: N = L%16, M = (L/16)*8 + i
  const int c16 = lane & 15, half = lane >> 4;
  #pragma unroll
  for (int nf = 0; nf < 4; ++nf) {
    const int col = colbase + wn * 64 + nf * 16 + c16;
    const float bv = bias[col];
    #pragma unroll
    for (int mf = 0; mf < 2; ++mf) {
      const size_t m0 = (size_t)(rowbase + wm * 32 + mf * 16 + half * 8);
      #pragma unroll
      for (int i = 0; i < 8; ++i) {
        float v = acc[mf][nf][i] + bv;
        if (OUT_F16) ((_Float16*)Cp)[(m0 + i) * 512 + col] = (_Float16)v;
        else         ((float*)Cp)[(m0 + i) * 512 + col]    = v;
      }
    }
  }
}

// ---------------------------------------------------------------------------
// kv accumulation: per (b,h), per 128-row chunk:
//   V = Xh @ W_v (WMMA), klog = Xh @ W_k_pad (WMMA), e = exp(klog)
//   kv[r][c] += sum_n e[n][r] * V[n][c]   (LDS reduce -> global f32 atomics)
// X chunk staged with async global->LDS DMA.
// ---------------------------------------------------------------------------
__global__ __launch_bounds__(256) void kvacc_k(
    const _Float16* __restrict__ xmid, const _Float16* __restrict__ Wtv,
    const _Float16* __restrict__ Wtk, float* __restrict__ kv_ws,
    float* __restrict__ ksum_ws) {
  __shared__ _Float16 X[128 * 72];
  __shared__ float e_lds[128 * 4];
  __shared__ float kv_red[256];
  __shared__ float ksum_red[4];
  const int tid  = threadIdx.x;
  const int lane = tid & 31;
  const int wid  = tid >> 5;
  const int bh   = blockIdx.y;
  const int b    = bh >> 3, h = bh & 7;
  const int row0 = b * NPERB + blockIdx.x * 128;

  kv_red[tid] = 0.f;
  if (tid < 4) ksum_red[tid] = 0.f;

  // stage Xh chunk: 128 x 64 f16 via async DMA
  #pragma unroll
  for (int i = 0; i < 4; ++i) {
    int q = tid + i * 256;
    int r = q >> 3, c8 = (q & 7) * 8;
    async_copy_b128(xmid + (size_t)(row0 + r) * 512 + h * 64 + c8,
                    &X[r * 72 + c8]);
  }
  wait_async0();
  __syncthreads();

  const int mrow = wid * 16;    // one 16-row A fragment per wave
  v8f vacc[4]; v8f kacc = vzero8();
  for (int nf = 0; nf < 4; ++nf) vacc[nf] = vzero8();

  #pragma unroll
  for (int ks = 0; ks < 64; ks += 32) {
    v16h a = load_a_lds(X, 72, mrow, ks, lane);
    #pragma unroll
    for (int nf = 0; nf < 4; ++nf) {
      v16h bf = load_b_gbl(Wtv, 64, nf * 16, ks, lane);
      vacc[nf] = wmma_f16(a, bf, vacc[nf]);
    }
    v16h bk = load_b_gbl(Wtk, 64, 0, ks, lane);
    kacc = wmma_f16(a, bk, kacc);
  }

  const int c16 = lane & 15, half = lane >> 4;
  if (c16 < 4) {
    #pragma unroll
    for (int i = 0; i < 8; ++i)
      e_lds[(mrow + half * 8 + i) * 4 + c16] = __expf(kacc[i]);
  }
  __syncthreads();

  // rank-4 accumulation over this chunk
  #pragma unroll
  for (int nf = 0; nf < 4; ++nf) {
    const int c = nf * 16 + c16;
    float kvl[4] = {0.f, 0.f, 0.f, 0.f};
    #pragma unroll
    for (int i = 0; i < 8; ++i) {
      const int r = mrow + half * 8 + i;
      const float vv = vacc[nf][i];
      #pragma unroll
      for (int rr = 0; rr < 4; ++rr) kvl[rr] += e_lds[r * 4 + rr] * vv;
    }
    #pragma unroll
    for (int rr = 0; rr < 4; ++rr) atomicAdd(&kv_red[rr * 64 + c], kvl[rr]);
  }
  {
    const int rr = tid & 3, rb = tid >> 2;   // 64 row-pairs x 4 r
    float s = e_lds[(rb * 2) * 4 + rr] + e_lds[(rb * 2 + 1) * 4 + rr];
    atomicAdd(&ksum_red[rr], s);
  }
  __syncthreads();

  atomicAdd(&kv_ws[bh * 256 + tid], kv_red[tid]);
  if (tid < 4) atomicAdd(&ksum_ws[bh * 4 + tid], ksum_red[tid]);
}

// ---------------------------------------------------------------------------
// qkv build: q = softmax_r(x_mid_h @ W_q); qkv = q @ (kv / ksum) -> f16
// block = 32 rows x 8 heads (256 tasks)
// ---------------------------------------------------------------------------
__global__ __launch_bounds__(256) void qkvbuild_k(
    const _Float16* __restrict__ xmid, const float* __restrict__ Wq,
    const float* __restrict__ kv_ws, const float* __restrict__ ksum_ws,
    _Float16* __restrict__ qkv) {
  __shared__ float kvn[8 * 4 * 64];
  __shared__ float wq[64 * 4];
  const int tid  = threadIdx.x;
  const int row0 = blockIdx.x * 32;
  const int b    = row0 >> 14;          // / 16384

  wq[tid] = Wq[tid];                    // 256 = 64*4
  #pragma unroll
  for (int i = 0; i < 8; ++i) {
    int q = tid + i * 256;              // 2048 = 8*4*64
    int hh = q >> 8; int rc = q & 255; int r = rc >> 6;
    kvn[q] = kv_ws[(size_t)(b * 8 + hh) * 256 + rc] /
             ksum_ws[(b * 8 + hh) * 4 + r];
  }
  __syncthreads();

  const int h = tid & 7;
  const int n = row0 + (tid >> 3);

  float xm[64];
  #pragma unroll
  for (int j = 0; j < 8; ++j) {
    v8h v = *(const v8h*)(xmid + (size_t)n * 512 + h * 64 + j * 8);
    #pragma unroll
    for (int t = 0; t < 8; ++t) xm[j * 8 + t] = (float)v[t];
  }
  float ql[4] = {0.f, 0.f, 0.f, 0.f};
  for (int d = 0; d < 64; ++d) {
    #pragma unroll
    for (int r = 0; r < 4; ++r) ql[r] += xm[d] * wq[d * 4 + r];
  }
  float mx = fmaxf(fmaxf(ql[0], ql[1]), fmaxf(ql[2], ql[3]));
  float s = 0.f;
  #pragma unroll
  for (int r = 0; r < 4; ++r) { ql[r] = __expf(ql[r] - mx); s += ql[r]; }
  const float inv = 1.f / s;
  #pragma unroll
  for (int r = 0; r < 4; ++r) ql[r] *= inv;

  const float* kvh = &kvn[h * 256];
  for (int c = 0; c < 64; ++c) {
    float o = ql[0] * kvh[c] + ql[1] * kvh[64 + c] +
              ql[2] * kvh[128 + c] + ql[3] * kvh[192 + c];
    qkv[(size_t)n * 512 + h * 64 + c] = (_Float16)o;
  }
}

// ---------------------------------------------------------------------------
// Host launcher
// ---------------------------------------------------------------------------
extern "C" void kernel_launch(void* const* d_in, const int* in_sizes, int n_in,
                              void* d_out, int out_size, void* d_ws,
                              size_t ws_size, hipStream_t stream) {
  (void)in_sizes; (void)n_in; (void)out_size; (void)ws_size;
  const float* x     = (const float*)d_in[0];
  const float* W_in  = (const float*)d_in[1];
  const float* b_in  = (const float*)d_in[2];
  const float* W_q   = (const float*)d_in[3];
  const float* W_k   = (const float*)d_in[4];
  const float* W_v   = (const float*)d_in[5];
  const float* W_out = (const float*)d_in[6];
  const float* b_out = (const float*)d_in[7];

  char* w = (char*)d_ws;
  _Float16* Wt_in  = (_Float16*)(w + 0);
  _Float16* Wt_out = (_Float16*)(w + 524288);
  _Float16* Wt_v   = (_Float16*)(w + 1048576);
  _Float16* Wt_k   = (_Float16*)(w + 1056768);
  float*    kv_ws  = (float*)   (w + 1058816);
  float*    ksum   = (float*)   (w + 1124352);
  _Float16* xmid   = (_Float16*)(w + 1125376);
  _Float16* qkv    = (_Float16*)(w + 1125376 + 134217728ull);

  // 1) weight prep + accumulator zeroing
  prep_k<<<(546048 + 255) / 256, 256, 0, stream>>>(
      W_in, W_out, W_v, W_k, Wt_in, Wt_out, Wt_v, Wt_k, kv_ws, ksum);

  // 2) x_mid = x @ W_in + b_in  (f16 out)
  gemm512_k<true, true><<<dim3(ROWS / 128, 4), 256, 0, stream>>>(
      (const void*)x, Wt_in, b_in, (void*)xmid);

  // 3) kv / ksum accumulation (k-softmax numerator/denominator + V proj)
  kvacc_k<<<dim3(NPERB / 128, HEADS * 8), 256, 0, stream>>>(
      xmid, Wt_v, Wt_k, kv_ws, ksum);

  // 4) qkv = softmax_r(q) @ (kv / ksum)  (f16 out)
  qkvbuild_k<<<ROWS / 32, 256, 0, stream>>>(xmid, W_q, kv_ws, ksum, qkv);

  // 5) out = qkv @ W_out + b_out  (f32 out)
  gemm512_k<false, false><<<dim3(ROWS / 128, 4), 256, 0, stream>>>(
      (const void*)qkv, Wt_out, b_out, d_out);
}
